// TIP_Attention_3685081940579
// MI455X (gfx1250) — matrix operand order repaired and checked by reading, not compiled, not run on hardware
//
#include <hip/hip_runtime.h>
#include <hip/hip_bf16.h>
#include <math.h>

#define B_    32
#define N_    1024
#define C_    768
#define NH_   12
#define HD_   64
#define KK_   9
#define PMD_  1536
#define CMD_  1536
#define HH_   32
#define WW_   32
#define TWO_C 1536
#define KDIM  768   // shared K dimension of both big GEMMs

typedef __attribute__((ext_vector_type(16))) __bf16 v16bf;
typedef __attribute__((ext_vector_type(8)))  float  v8f;

__device__ __forceinline__ unsigned short f2bf(float f) {
  unsigned u = __float_as_uint(f);
  u += 0x7FFFu + ((u >> 16) & 1u);           // round-to-nearest-even
  return (unsigned short)(u >> 16);
}

__device__ __forceinline__ float gelu_exact(float x) {
  return 0.5f * x * (1.0f + erff(x * 0.70710678118654752440f));
}

// ---------------- prep kernels ----------------

__global__ void k_f32_to_bf16(const float* __restrict__ src,
                              unsigned short* __restrict__ dst, int n) {
  int i = blockIdx.x * blockDim.x + threadIdx.x;
  if (i < n) dst[i] = f2bf(src[i]);
}

// W: [K][N] fp32  ->  WT: [N][K] bf16
__global__ void k_transpose_f2bf(const float* __restrict__ W,
                                 unsigned short* __restrict__ WT, int K, int N) {
  int i = blockIdx.x * blockDim.x + threadIdx.x;
  if (i >= K * N) return;
  int n = i / K, k = i - n * K;
  WT[i] = f2bf(W[(size_t)k * N + n]);
}

// per-row LayerNorm (rows x ncols), 256 threads/row
__global__ __launch_bounds__(256) void k_layernorm_rows(
    const float* __restrict__ X, float* __restrict__ Y,
    const float* __restrict__ g, const float* __restrict__ b,
    int ncols, float eps) {
  __shared__ float red[256];
  int row = blockIdx.x, tid = threadIdx.x;
  const float* x = X + (size_t)row * ncols;
  float s = 0.f;
  for (int c = tid; c < ncols; c += 256) s += x[c];
  red[tid] = s; __syncthreads();
  for (int o = 128; o > 0; o >>= 1) { if (tid < o) red[tid] += red[tid + o]; __syncthreads(); }
  float mean = red[0] / ncols;
  __syncthreads();
  s = 0.f;
  for (int c = tid; c < ncols; c += 256) { float d = x[c] - mean; s += d * d; }
  red[tid] = s; __syncthreads();
  for (int o = 128; o > 0; o >>= 1) { if (tid < o) red[tid] += red[tid + o]; __syncthreads(); }
  float rstd = rsqrtf(red[0] / ncols + eps);
  for (int c = tid; c < ncols; c += 256)
    Y[(size_t)row * ncols + c] = (x[c] - mean) * rstd * g[c] + b[c];
}

// ---------------- WMMA bf16 GEMM: C[M][N] = A[M][KDIM] * WT[N][KDIM]^T ----------------
// block = 256 threads = 8 waves; wave w: rows [bx*128 + w*16, +16), cols [by*64, +64).
// K-loop fully unrolled with register double-buffered fragments so next-step
// loads are issued before the current step's WMMAs (hides VMEM latency).
struct bfrag { union { v16bf v; uint4 u[2]; }; };

__global__ __launch_bounds__(256) void k_gemm_bf16(
    const unsigned short* __restrict__ A,    // [M][KDIM] bf16 bits
    const unsigned short* __restrict__ WT,   // [N][KDIM] bf16 bits
    float* __restrict__ C, int N) {
  const int lane = threadIdx.x & 31;
  const int wave = threadIdx.x >> 5;
  const int m0 = blockIdx.x * 128 + wave * 16;
  const int n0 = blockIdx.y * 64;
  v8f acc[4] = {};
  const int mrow = m0 + (lane & 15);
  const int kgA  = (lane < 16) ? 0 : 16;    // A frag: K = kgA + h, 16 contiguous halfs
  const int ncol = lane & 15;
  const int kgB  = (lane < 16) ? 0 : 16;    // B frag: K = kgB + h, 16 contiguous halfs
  const unsigned short* pa  = A  + (size_t)mrow * KDIM + kgA;
  const unsigned short* pb0 = WT + (size_t)(n0 + ncol) * KDIM + kgB;

  bfrag af[2];
  bfrag bf[2][4];

  // prologue: load k-step 0 into buffer 0
  af[0].u[0] = *(const uint4*)(pa);
  af[0].u[1] = *(const uint4*)(pa + 8);
#pragma unroll
  for (int j = 0; j < 4; ++j) {
    const unsigned short* pb = pb0 + (size_t)(j * 16) * KDIM;
    bf[0][j].u[0] = *(const uint4*)(pb);
    bf[0][j].u[1] = *(const uint4*)(pb + 8);
  }

#pragma unroll
  for (int k0 = 0; k0 < KDIM; k0 += 32) {
    const int cur = (k0 >> 5) & 1;
    const int nxt = cur ^ 1;
    const int kn = k0 + 32;
    if (kn < KDIM) {
      af[nxt].u[0] = *(const uint4*)(pa + kn);
      af[nxt].u[1] = *(const uint4*)(pa + kn + 8);
#pragma unroll
      for (int j = 0; j < 4; ++j) {
        const unsigned short* pb = pb0 + (size_t)(j * 16) * KDIM + kn;
        bf[nxt][j].u[0] = *(const uint4*)(pb);
        bf[nxt][j].u[1] = *(const uint4*)(pb + 8);
      }
    }
#pragma unroll
    for (int j = 0; j < 4; ++j)
      acc[j] = __builtin_amdgcn_wmma_f32_16x16x32_bf16(
          false, af[cur].v, false, bf[cur][j].v, (short)0, acc[j], false, false);
  }

  const int mb = m0 + ((lane < 16) ? 0 : 8);
#pragma unroll
  for (int j = 0; j < 4; ++j) {
    int col = n0 + j * 16 + ncol;
#pragma unroll
    for (int r = 0; r < 8; ++r)
      C[(size_t)(mb + r) * N + col] = acc[j][r];
  }
}

// ---------------- attention: per (b,h), 9 queries over 1024 keys ----------------
__global__ __launch_bounds__(256) void k_attention(
    const float* __restrict__ kv,    // [B*N][1536] (k: cols 0..767 by head, v: +768)
    const float* __restrict__ wmq,   // [9][768] normalized queries
    float* __restrict__ ca) {        // [B][9][768]
  __shared__ float s[KK_][N_];       // 36864 B
  __shared__ float qs[KK_][HD_];     // 2304 B
  __shared__ float vblk[64][HD_];    // 16384 B
  __shared__ float red[256];
  const int b = blockIdx.x / NH_;
  const int h = blockIdx.x % NH_;
  const int tid = threadIdx.x;
  for (int i = tid; i < KK_ * HD_; i += 256)
    qs[i / HD_][i % HD_] = wmq[(i / HD_) * C_ + h * HD_ + (i % HD_)];
  __syncthreads();
  // logits
  for (int n = tid; n < N_; n += 256) {
    const float* krow = kv + (size_t)(b * N_ + n) * TWO_C + h * HD_;
    float acc[KK_] = {0.f};
    for (int d = 0; d < HD_; d += 4) {
      float4 k4 = *(const float4*)(krow + d);
#pragma unroll
      for (int q = 0; q < KK_; ++q)
        acc[q] += qs[q][d] * k4.x + qs[q][d + 1] * k4.y +
                  qs[q][d + 2] * k4.z + qs[q][d + 3] * k4.w;
    }
#pragma unroll
    for (int q = 0; q < KK_; ++q) s[q][n] = acc[q] * 0.125f;  // HD^-0.5
  }
  __syncthreads();
  // softmax per row
  for (int q = 0; q < KK_; ++q) {
    float mx = -3.402823466e38f;
    for (int n = tid; n < N_; n += 256) mx = fmaxf(mx, s[q][n]);
    red[tid] = mx; __syncthreads();
    for (int o = 128; o > 0; o >>= 1) { if (tid < o) red[tid] = fmaxf(red[tid], red[tid + o]); __syncthreads(); }
    mx = red[0]; __syncthreads();
    float sum = 0.f;
    for (int n = tid; n < N_; n += 256) { float e = expf(s[q][n] - mx); s[q][n] = e; sum += e; }
    red[tid] = sum; __syncthreads();
    for (int o = 128; o > 0; o >>= 1) { if (tid < o) red[tid] += red[tid + o]; __syncthreads(); }
    float inv = 1.f / red[0]; __syncthreads();
    for (int n = tid; n < N_; n += 256) s[q][n] *= inv;
  }
  __syncthreads();
  // O = P @ V, 576 outputs -> threads own t, t+256, t+512
  float o0 = 0.f, o1 = 0.f, o2 = 0.f;
  const int t0 = tid, t1 = tid + 256, t2 = tid + 512;
  for (int nc = 0; nc < N_; nc += 64) {
    for (int i = tid; i < 64 * HD_; i += 256) {
      int nn = i >> 6, dd = i & 63;
      vblk[nn][dd] = kv[(size_t)(b * N_ + nc + nn) * TWO_C + C_ + h * HD_ + dd];
    }
    __syncthreads();
    { int m = t0 / HD_, d = t0 % HD_; for (int n = 0; n < 64; ++n) o0 += s[m][nc + n] * vblk[n][d]; }
    { int m = t1 / HD_, d = t1 % HD_; for (int n = 0; n < 64; ++n) o1 += s[m][nc + n] * vblk[n][d]; }
    if (t2 < KK_ * HD_) { int m = t2 / HD_, d = t2 % HD_; for (int n = 0; n < 64; ++n) o2 += s[m][nc + n] * vblk[n][d]; }
    __syncthreads();
  }
  { int m = t0 / HD_, d = t0 % HD_; ca[((size_t)b * KK_ + m) * C_ + h * HD_ + d] = o0; }
  { int m = t1 / HD_, d = t1 % HD_; ca[((size_t)b * KK_ + m) * C_ + h * HD_ + d] = o1; }
  if (t2 < KK_ * HD_) { int m = t2 / HD_, d = t2 % HD_; ca[((size_t)b * KK_ + m) * C_ + h * HD_ + d] = o2; }
}

// ---------------- w1 = kernel + ca @ proj_w + proj_b ----------------
__global__ __launch_bounds__(256) void k_proj_res(
    const float* __restrict__ ca, const float* __restrict__ projw,
    const float* __restrict__ projb, const float* __restrict__ kern,
    float* __restrict__ w1) {
  __shared__ float car[C_];
  const int bq = blockIdx.x / 3;
  const int cpart = blockIdx.x % 3;
  const float* crow = ca + (size_t)bq * C_;
  for (int i = threadIdx.x; i < C_; i += 256) car[i] = crow[i];
  __syncthreads();
  const int q = bq % KK_;
  const int c = cpart * 256 + threadIdx.x;
  float acc = projb[c] + kern[q * C_ + c];
  for (int k = 0; k < C_; ++k) acc += car[k] * projw[(size_t)k * C_ + c];
  w1[(size_t)bq * C_ + c] = acc;
}

// ---------------- patch mix: w1 += (gelu(LN1(w1)^T @ pm1 + b1) @ pm2 + b2)^T ----------------
__global__ __launch_bounds__(256) void k_patchmix(
    const float* __restrict__ tln, const float* __restrict__ pm1w,
    const float* __restrict__ pm1b, const float* __restrict__ pm2w,
    const float* __restrict__ pm2b, float* __restrict__ w1) {
  const int i = blockIdx.x * 256 + threadIdx.x;   // i < B*C
  const int b = i / C_, c = i % C_;
  float tq[KK_];
#pragma unroll
  for (int q = 0; q < KK_; ++q) tq[q] = tln[((size_t)b * KK_ + q) * C_ + c];
  float acc[KK_] = {0.f};
  for (int j = 0; j < PMD_; ++j) {
    float sum = pm1b[j];
#pragma unroll
    for (int q = 0; q < KK_; ++q) sum += tq[q] * pm1w[q * PMD_ + j];
    float g = gelu_exact(sum);
#pragma unroll
    for (int q = 0; q < KK_; ++q) acc[q] += g * pm2w[j * KK_ + q];
  }
#pragma unroll
  for (int q = 0; q < KK_; ++q)
    w1[((size_t)b * KK_ + q) * C_ + c] += acc[q] + pm2b[q];
}

// ---------------- channel mix: w1 += gelu(LN2(w1) @ cm1 + b1) @ cm2 + b2 ----------------
__global__ __launch_bounds__(256) void k_channelmix(
    const float* __restrict__ n2g, const float* __restrict__ n2b,
    const float* __restrict__ cm1w, const float* __restrict__ cm1b,
    const float* __restrict__ cm2w, const float* __restrict__ cm2b,
    float* __restrict__ w1) {
  __shared__ float t[C_];
  __shared__ float g[CMD_];
  __shared__ float red[256];
  const int bq = blockIdx.x, tid = threadIdx.x;
  float* row = w1 + (size_t)bq * C_;
  float ssum = 0.f;
  for (int c = tid; c < C_; c += 256) ssum += row[c];
  red[tid] = ssum; __syncthreads();
  for (int o = 128; o > 0; o >>= 1) { if (tid < o) red[tid] += red[tid + o]; __syncthreads(); }
  float mean = red[0] / C_; __syncthreads();
  ssum = 0.f;
  for (int c = tid; c < C_; c += 256) { float d = row[c] - mean; ssum += d * d; }
  red[tid] = ssum; __syncthreads();
  for (int o = 128; o > 0; o >>= 1) { if (tid < o) red[tid] += red[tid + o]; __syncthreads(); }
  float rstd = rsqrtf(red[0] / C_ + 1e-5f);
  for (int c = tid; c < C_; c += 256) t[c] = (row[c] - mean) * rstd * n2g[c] + n2b[c];
  __syncthreads();
  for (int j = tid; j < CMD_; j += 256) {
    float s = cm1b[j];
    for (int k = 0; k < C_; ++k) s += t[k] * cm1w[(size_t)k * CMD_ + j];
    g[j] = gelu_exact(s);
  }
  __syncthreads();
  for (int c = tid; c < C_; c += 256) {
    float s = cm2b[c];
    for (int j = 0; j < CMD_; ++j) s += g[j] * cm2w[(size_t)j * C_ + c];
    row[c] += s;
  }
}

// ---------------- dynamic per-(b,c) 3x3 depthwise conv, all in [B,N,C] layout ----------------
__global__ __launch_bounds__(256) void k_dynconv(
    const float* __restrict__ xq, const float* __restrict__ w1,
    float* __restrict__ y) {
  const int bn = blockIdx.x;
  const int b = bn >> 10;
  const int n = bn & 1023;
  const int h = n >> 5, w = n & 31;
  const float* fbase = w1 + (size_t)b * KK_ * C_;
#pragma unroll
  for (int p = 0; p < 3; ++p) {
    const int c = p * 256 + threadIdx.x;
    float acc = 0.f;
#pragma unroll
    for (int dh = -1; dh <= 1; ++dh) {
      const int hh = h + dh;
      if (hh < 0 || hh >= HH_) continue;
#pragma unroll
      for (int dw = -1; dw <= 1; ++dw) {
        const int ww = w + dw;
        if (ww < 0 || ww >= WW_) continue;
        const int k = (dh + 1) * 3 + (dw + 1);
        acc += fbase[(size_t)k * C_ + c] *
               xq[((size_t)b * N_ + hh * WW_ + ww) * C_ + c];
      }
    }
    y[((size_t)b * N_ + n) * C_ + c] = acc;
  }
}

// ---------------- launch ----------------

extern "C" void kernel_launch(void* const* d_in, const int* in_sizes, int n_in,
                              void* d_out, int out_size, void* d_ws, size_t ws_size,
                              hipStream_t stream) {
  (void)in_sizes; (void)n_in; (void)out_size; (void)ws_size;
  const float* x      = (const float*)d_in[0];
  const float* kern   = (const float*)d_in[3];
  const float* ln_g   = (const float*)d_in[4];
  const float* ln_b   = (const float*)d_in[5];
  const float* kv_w   = (const float*)d_in[6];
  const float* proj_w = (const float*)d_in[7];
  const float* proj_b = (const float*)d_in[8];
  const float* pq_w   = (const float*)d_in[9];
  const float* n1g    = (const float*)d_in[10];
  const float* n1b    = (const float*)d_in[11];
  const float* n2g    = (const float*)d_in[12];
  const float* n2b    = (const float*)d_in[13];
  const float* pm1w   = (const float*)d_in[14];
  const float* pm1b   = (const float*)d_in[15];
  const float* pm2w   = (const float*)d_in[16];
  const float* pm2b   = (const float*)d_in[17];
  const float* cm1w   = (const float*)d_in[18];
  const float* cm1b   = (const float*)d_in[19];
  const float* cm2w   = (const float*)d_in[20];
  const float* cm2b   = (const float*)d_in[21];
  float* y = (float*)d_out;

  char* ws = (char*)d_ws;
  size_t off = 0;
  auto alloc = [&](size_t bytes) {
    void* p = ws + off;
    off = (off + bytes + 255) & ~(size_t)255;
    return p;
  };
  unsigned short* xh  = (unsigned short*)alloc((size_t)B_ * N_ * C_ * 2);
  unsigned short* kvT = (unsigned short*)alloc((size_t)TWO_C * C_ * 2);
  unsigned short* pqT = (unsigned short*)alloc((size_t)C_ * C_ * 2);
  float* wmq = (float*)alloc((size_t)KK_ * C_ * 4);
  float* kv  = (float*)alloc((size_t)B_ * N_ * TWO_C * 4);
  float* ca  = (float*)alloc((size_t)B_ * KK_ * C_ * 4);
  float* w1  = (float*)alloc((size_t)B_ * KK_ * C_ * 4);
  float* tln = (float*)alloc((size_t)B_ * KK_ * C_ * 4);
  float* xq  = (float*)alloc((size_t)B_ * N_ * C_ * 4);

  const int nx = B_ * N_ * C_;
  k_f32_to_bf16<<<(nx + 255) / 256, 256, 0, stream>>>(x, xh, nx);
  k_transpose_f2bf<<<(C_ * TWO_C + 255) / 256, 256, 0, stream>>>(kv_w, kvT, C_, TWO_C);
  k_transpose_f2bf<<<(C_ * C_ + 255) / 256, 256, 0, stream>>>(pq_w, pqT, C_, C_);
  k_layernorm_rows<<<KK_, 256, 0, stream>>>(kern, wmq, ln_g, ln_b, C_, 1e-6f);
  k_gemm_bf16<<<dim3(B_ * N_ / 128, TWO_C / 64), 256, 0, stream>>>(xh, kvT, kv, TWO_C);
  k_attention<<<B_ * NH_, 256, 0, stream>>>(kv, wmq, ca);
  k_proj_res<<<B_ * KK_ * 3, 256, 0, stream>>>(ca, proj_w, proj_b, kern, w1);
  k_layernorm_rows<<<B_ * KK_, 256, 0, stream>>>(w1, tln, n1g, n1b, C_, 1e-5f);
  k_patchmix<<<B_ * C_ / 256, 256, 0, stream>>>(tln, pm1w, pm1b, pm2w, pm2b, w1);
  k_channelmix<<<B_ * KK_, 256, 0, stream>>>(n2g, n2b, cm1w, cm1b, cm2w, cm2b, w1);
  k_gemm_bf16<<<dim3(B_ * N_ / 128, C_ / 64), 256, 0, stream>>>(xh, pqT, xq, C_);
  k_dynconv<<<B_ * N_, 256, 0, stream>>>(xq, w1, y);
}
